// CrossCondGPT2_35682588295188
// MI455X (gfx1250) — compile-verified, matrix operand order checked by reading, not run on hardware
//
#include <hip/hip_runtime.h>
#include <hip/hip_bf16.h>
#include <math.h>

// ---------------------------------------------------------------------------
// CrossCondGPT2 forward for gfx1250 (MI455X).
// bf16 WMMA (v_wmma_f32_16x16x32_bf16) for all GEMMs + flash attention.
// Residual stream kept in fp32 in d_out. Weights converted+transposed to
// bf16 [N][K] in d_ws each call (deterministic).
// GEMM wave tile: 32(M) x 64(N)  -> 8 WMMAs per 32-K step sharing B frags.
// Workspace requirement: ~120 MB.
// ---------------------------------------------------------------------------

typedef __bf16 bf16;
typedef __attribute__((ext_vector_type(16))) __bf16 v16bf;
typedef __attribute__((ext_vector_type(8)))  __bf16 v8bf;
typedef __attribute__((ext_vector_type(8)))  float  v8f;

#define DEV static __device__ __forceinline__

// Load a 16x32 (or 32x16) bf16 WMMA fragment for one lane.
// `base` points at the 32-element K-run for this lane's row/column.
// Per ISA layout: lane-half hf holds K = 8*hf+(0..7) in elems 0..7 and
// K = 16+8*hf+(0..7) in elems 8..15 -> two contiguous 16B loads.
DEV v16bf load_frag(const bf16* base, int hf) {
  union { v16bf v; v8bf h[2]; } u;
  u.h[0] = *reinterpret_cast<const v8bf*>(base + 8 * hf);
  u.h[1] = *reinterpret_cast<const v8bf*>(base + 16 + 8 * hf);
  return u.v;
}

DEV v8f wmma_bf16(v16bf a, v16bf b, v8f c) {
  return __builtin_amdgcn_wmma_f32_16x16x32_bf16(
      /*neg_a=*/false, a, /*neg_b=*/false, b,
      /*c_mod=*/(short)0, c, /*reuse_a=*/false, /*reuse_b=*/false);
}

// ---------------------------------------------------------------------------
// Weight convert + transpose: src fp32 [L][K][N] -> dst bf16 [L][N][K]
// ---------------------------------------------------------------------------
__global__ void convT_kernel(const float* __restrict__ src, bf16* __restrict__ dst,
                             int K, int N) {
  long l = blockIdx.y;
  long idx = (long)blockIdx.x * 256 + threadIdx.x;
  long kn = (long)K * N;
  if (idx < kn) {
    long kk = idx / N, nn = idx - kk * (long)N;
    dst[l * kn + nn * (long)K + kk] = (bf16)src[l * kn + idx];
  }
}

__global__ void copy_f32_kernel(const float* __restrict__ s, float* __restrict__ d, long n) {
  long i = (long)blockIdx.x * 256 + threadIdx.x;
  if (i < n) d[i] = s[i];
}

// ---------------------------------------------------------------------------
// LayerNorm: fp32 residual row (768) -> bf16 normalized row
// one block (256 threads) per token
// ---------------------------------------------------------------------------
__global__ __launch_bounds__(256) void ln_kernel(const float* __restrict__ x,
                                                 const float* __restrict__ g,
                                                 const float* __restrict__ bb,
                                                 bf16* __restrict__ out) {
  __shared__ float sA[256], sB[256];
  int tid = threadIdx.x;
  const float* xr = x + (size_t)blockIdx.x * 768;
  float s = 0.f, ss = 0.f;
  for (int i = tid; i < 768; i += 256) { float v = xr[i]; s += v; ss += v * v; }
  sA[tid] = s; sB[tid] = ss;
  __syncthreads();
  for (int o = 128; o > 0; o >>= 1) {
    if (tid < o) { sA[tid] += sA[tid + o]; sB[tid] += sB[tid + o]; }
    __syncthreads();
  }
  float mu = sA[0] * (1.0f / 768.0f);
  float var = sB[0] * (1.0f / 768.0f) - mu * mu;
  float rs = rsqrtf(var + 1e-5f);
  bf16* orow = out + (size_t)blockIdx.x * 768;
  for (int i = tid; i < 768; i += 256)
    orow[i] = (bf16)((xr[i] - mu) * rs * g[i] + bb[i]);
}

// ---------------------------------------------------------------------------
// bf16 WMMA GEMM: out[M,N] = A[M,K] @ Bt[N,K]^T + bias
//   mode 0: store bf16 row-major
//   mode 1: store bf16 row-major with exact GELU
//   mode 2: store bf16 transposed-per-head [b][h][d][t] (for V)
//   mode 3: fp32 residual accumulate: resid[m,n] += val
// 256 threads = 8 waves (2 along M x 4 along N).
// Wave tile 32(M) x 64(N); block tile 64 x 256.
// ---------------------------------------------------------------------------
__global__ __launch_bounds__(256) void gemm_bf16_kernel(
    const bf16* __restrict__ A, const bf16* __restrict__ Bt,
    const float* __restrict__ bias, bf16* __restrict__ outb,
    float* __restrict__ resid, int Md, int Nd, int Kd, int mode) {
  int lane = threadIdx.x & 31;
  int wave = threadIdx.x >> 5;
  int wn = wave & 3, wm = wave >> 2;
  int m0 = blockIdx.y * 64 + wm * 32;
  int n0 = blockIdx.x * 256 + wn * 64;
  int lm = lane & 15, hf = lane >> 4;

  const v8f vz = {0.f, 0.f, 0.f, 0.f, 0.f, 0.f, 0.f, 0.f};
  v8f acc[8] = {vz, vz, vz, vz, vz, vz, vz, vz};  // [m-half 0..1][n-tile 0..3]

  const bf16* arow0 = A + (size_t)(m0 + lm) * Kd;        // rows m0   .. m0+15
  const bf16* arow1 = A + (size_t)(m0 + 16 + lm) * Kd;   // rows m0+16.. m0+31
  const bf16* b0 = Bt + (size_t)(n0 + 0 * 16 + lm) * Kd;
  const bf16* b1 = Bt + (size_t)(n0 + 1 * 16 + lm) * Kd;
  const bf16* b2 = Bt + (size_t)(n0 + 2 * 16 + lm) * Kd;
  const bf16* b3 = Bt + (size_t)(n0 + 3 * 16 + lm) * Kd;

  for (int k0 = 0; k0 < Kd; k0 += 32) {
    v16bf a0 = load_frag(arow0 + k0, hf);
    v16bf a1 = load_frag(arow1 + k0, hf);
    v16bf bt0 = load_frag(b0 + k0, hf);
    acc[0] = wmma_bf16(a0, bt0, acc[0]);
    acc[4] = wmma_bf16(a1, bt0, acc[4]);
    v16bf bt1 = load_frag(b1 + k0, hf);
    acc[1] = wmma_bf16(a0, bt1, acc[1]);
    acc[5] = wmma_bf16(a1, bt1, acc[5]);
    v16bf bt2 = load_frag(b2 + k0, hf);
    acc[2] = wmma_bf16(a0, bt2, acc[2]);
    acc[6] = wmma_bf16(a1, bt2, acc[6]);
    v16bf bt3 = load_frag(b3 + k0, hf);
    acc[3] = wmma_bf16(a0, bt3, acc[3]);
    acc[7] = wmma_bf16(a1, bt3, acc[7]);
  }

#pragma unroll
  for (int t = 0; t < 4; ++t) {
    int n = n0 + t * 16 + lm;
    float bvv = bias[n];
#pragma unroll
    for (int mh = 0; mh < 2; ++mh) {
#pragma unroll
      for (int r = 0; r < 8; ++r) {
        int m = m0 + mh * 16 + 8 * hf + r;  // C-layout: lanes 0-15 M=r, 16-31 M=8+r
        float v = acc[mh * 4 + t][r] + bvv;
        if (mode == 1) v = 0.5f * v * (1.0f + erff(v * 0.70710678118654752f));
        if (mode == 3) {
          resid[(size_t)m * Nd + n] += v;
        } else if (mode == 2) {
          int bb = m >> 10, tp = m & 1023;   // token -> (batch, t)
          int hh = n >> 6,  dd = n & 63;     // channel -> (head, d)
          outb[(((size_t)bb * 12 + hh) * 64 + dd) * 1024 + tp] = (bf16)v;
        } else {
          outb[(size_t)m * Nd + n] = (bf16)v;
        }
      }
    }
  }
}

// ---------------------------------------------------------------------------
// Flash attention, one wave per (b, h, 16-query tile).
// q,k: bf16 [B,T,C] row-major. vT: bf16 [B,NH,64,T]. y: bf16 [B,T,C].
// mask[q,k] = (q%256) >= (k%256)  (tril(256) tiled 4x4)
// ---------------------------------------------------------------------------
__global__ __launch_bounds__(256) void attn_kernel(
    const bf16* __restrict__ q, const bf16* __restrict__ k,
    const bf16* __restrict__ vT, bf16* __restrict__ y) {
  __shared__ __align__(16) bf16 pl[8][16][32];  // per-wave P tile (16x32)

  int lane = threadIdx.x & 31;
  int wave = threadIdx.x >> 5;
  int gw = blockIdx.x * 8 + wave;            // 0 .. 3071
  int qt = gw & 63;                          // query tile
  int h  = (gw >> 6) % 12;
  int b  = gw / (64 * 12);
  int lm = lane & 15, hf = lane >> 4;
  const float scale = 0.125f;                // 1/sqrt(64)

  // Q fragments (16x32 each, d 0..31 and 32..63), live in regs all loop
  const bf16* qrow = q + ((size_t)(b * 1024 + qt * 16 + lm)) * 768 + h * 64;
  v16bf qa0 = load_frag(qrow, hf);
  v16bf qa1 = load_frag(qrow + 32, hf);

  const v8f vzv = {0.f, 0.f, 0.f, 0.f, 0.f, 0.f, 0.f, 0.f};
  v8f o[4] = {vzv, vzv, vzv, vzv};
  float mi[8], li[8];
#pragma unroll
  for (int r = 0; r < 8; ++r) { mi[r] = -1e30f; li[r] = 0.f; }

  const bf16* kbase = k + ((size_t)b * 1024) * 768 + h * 64;
  const bf16* vbase = vT + (((size_t)b * 12 + h) * 64) * 1024;

  for (int kb = 0; kb < 1024; kb += 32) {
    // ---- S = Q @ K^T : two 16x16 tiles (keys kb..kb+15, kb+16..kb+31)
    v8f s0 = vzv, s1 = vzv;
    const bf16* kr0 = kbase + (size_t)(kb + lm) * 768;
    const bf16* kr1 = kbase + (size_t)(kb + 16 + lm) * 768;
    s0 = wmma_bf16(qa0, load_frag(kr0, hf), s0);
    s0 = wmma_bf16(qa1, load_frag(kr0 + 32, hf), s0);
    s1 = wmma_bf16(qa0, load_frag(kr1, hf), s1);
    s1 = wmma_bf16(qa1, load_frag(kr1 + 32, hf), s1);

    // ---- scale + mask + online softmax
    int n0m = (kb + lm) & 255;
    int n1m = (kb + 16 + lm) & 255;
    float p0[8], p1[8], rm[8];
#pragma unroll
    for (int r = 0; r < 8; ++r) {
      int im = (qt * 16 + 8 * hf + r) & 255;
      float x0 = s0[r] * scale; if (n0m > im) x0 = -1e30f;
      float x1 = s1[r] * scale; if (n1m > im) x1 = -1e30f;
      p0[r] = x0; p1[r] = x1;
      rm[r] = fmaxf(x0, x1);
    }
#pragma unroll
    for (int off = 1; off < 16; off <<= 1)
#pragma unroll
      for (int r = 0; r < 8; ++r)
        rm[r] = fmaxf(rm[r], __shfl_xor(rm[r], off, 16));

    float rs[8], al[8];
#pragma unroll
    for (int r = 0; r < 8; ++r) {
      float mn = fmaxf(mi[r], rm[r]);
      al[r] = __expf(mi[r] - mn);
      float e0 = __expf(p0[r] - mn);
      float e1 = __expf(p1[r] - mn);
      p0[r] = e0; p1[r] = e1;
      rs[r] = e0 + e1;
      mi[r] = mn;
    }
#pragma unroll
    for (int off = 1; off < 16; off <<= 1)
#pragma unroll
      for (int r = 0; r < 8; ++r)
        rs[r] += __shfl_xor(rs[r], off, 16);
#pragma unroll
    for (int r = 0; r < 8; ++r) li[r] = li[r] * al[r] + rs[r];
#pragma unroll
    for (int t = 0; t < 4; ++t)
#pragma unroll
      for (int r = 0; r < 8; ++r) o[t][r] *= al[r];

    // ---- P (C-layout) -> LDS -> A-layout fragment
#pragma unroll
    for (int r = 0; r < 8; ++r) {
      pl[wave][8 * hf + r][lm]      = (bf16)p0[r];
      pl[wave][8 * hf + r][16 + lm] = (bf16)p1[r];
    }
    v16bf pa = load_frag(&pl[wave][lm][0], hf);

    // ---- O += P @ V  (V pre-transposed: rows contiguous in key)
#pragma unroll
    for (int dt = 0; dt < 4; ++dt) {
      const bf16* vr = vbase + (size_t)(dt * 16 + lm) * 1024 + kb;
      o[dt] = wmma_bf16(pa, load_frag(vr, hf), o[dt]);
    }
  }

  // ---- normalize and store y
#pragma unroll
  for (int dt = 0; dt < 4; ++dt)
#pragma unroll
    for (int r = 0; r < 8; ++r) {
      float val = o[dt][r] / li[r];
      size_t m = (size_t)(b * 1024 + qt * 16 + 8 * hf + r);
      y[m * 768 + h * 64 + dt * 16 + lm] = (bf16)val;
    }
}

// ---------------------------------------------------------------------------
// Host-side orchestration
// ---------------------------------------------------------------------------
extern "C" void kernel_launch(void* const* d_in, const int* in_sizes, int n_in,
                              void* d_out, int out_size, void* d_ws, size_t ws_size,
                              hipStream_t stream) {
  (void)in_sizes; (void)n_in; (void)out_size; (void)ws_size;
  const float* x    = (const float*)d_in[0];
  const float* ln1g = (const float*)d_in[1];
  const float* ln1b = (const float*)d_in[2];
  const float* Wq   = (const float*)d_in[3];
  const float* bq   = (const float*)d_in[4];
  const float* Wk   = (const float*)d_in[5];
  const float* bk   = (const float*)d_in[6];
  const float* Wv   = (const float*)d_in[7];
  const float* bv   = (const float*)d_in[8];
  const float* Wp   = (const float*)d_in[9];
  const float* bp   = (const float*)d_in[10];
  const float* ln2g = (const float*)d_in[11];
  const float* ln2b = (const float*)d_in[12];
  const float* W1   = (const float*)d_in[13];
  const float* b1   = (const float*)d_in[14];
  const float* W2   = (const float*)d_in[15];
  const float* b2   = (const float*)d_in[16];

  float* h = (float*)d_out;  // fp32 residual stream lives in the output buffer

  const long SW  = 768L * 768;    // per-layer 768x768 weight
  const long SW1 = 768L * 3072;   // per-layer 768x3072 / 3072x768 weight
  const long ACT = 4096L * 768;
  const long FF  = 4096L * 3072;

  bf16* p = (bf16*)d_ws;
  bf16* WqT = p; p += 4 * SW;
  bf16* WkT = p; p += 4 * SW;
  bf16* WvT = p; p += 4 * SW;
  bf16* WpT = p; p += 4 * SW;
  bf16* W1T = p; p += 4 * SW1;
  bf16* W2T = p; p += 4 * SW1;
  bf16* a_in = p; p += ACT;
  bf16* qb   = p; p += ACT;
  bf16* kb   = p; p += ACT;
  bf16* vTb  = p; p += ACT;
  bf16* yb   = p; p += ACT;
  bf16* m_in = p; p += ACT;
  bf16* ffb  = p; p += FF;

  // --- convert + transpose all weights to bf16 [N][K]
  {
    dim3 g((unsigned)((SW + 255) / 256), 4);
    convT_kernel<<<g, 256, 0, stream>>>(Wq, WqT, 768, 768);
    convT_kernel<<<g, 256, 0, stream>>>(Wk, WkT, 768, 768);
    convT_kernel<<<g, 256, 0, stream>>>(Wv, WvT, 768, 768);
    convT_kernel<<<g, 256, 0, stream>>>(Wp, WpT, 768, 768);
    dim3 g1((unsigned)((SW1 + 255) / 256), 4);
    convT_kernel<<<g1, 256, 0, stream>>>(W1, W1T, 768, 3072);
    convT_kernel<<<g1, 256, 0, stream>>>(W2, W2T, 3072, 768);
  }

  // --- residual init: h = x
  copy_f32_kernel<<<(unsigned)((ACT + 255) / 256), 256, 0, stream>>>(x, h, ACT);

  const dim3 gC(3, 64);    // N=768  : 768/256 x 4096/64
  const dim3 gH(12, 64);   // N=3072 : 3072/256 x 4096/64
  for (int l = 0; l < 4; ++l) {
    // attention branch
    ln_kernel<<<4096, 256, 0, stream>>>(h, ln1g + l * 768, ln1b + l * 768, a_in);
    gemm_bf16_kernel<<<gC, 256, 0, stream>>>(a_in, WqT + (long)l * SW, bq + l * 768,
                                             qb, nullptr, 4096, 768, 768, 0);
    gemm_bf16_kernel<<<gC, 256, 0, stream>>>(a_in, WkT + (long)l * SW, bk + l * 768,
                                             kb, nullptr, 4096, 768, 768, 0);
    gemm_bf16_kernel<<<gC, 256, 0, stream>>>(a_in, WvT + (long)l * SW, bv + l * 768,
                                             vTb, nullptr, 4096, 768, 768, 2);
    attn_kernel<<<384, 256, 0, stream>>>(qb, kb, vTb, yb);
    gemm_bf16_kernel<<<gC, 256, 0, stream>>>(yb, WpT + (long)l * SW, bp + l * 768,
                                             nullptr, h, 4096, 768, 768, 3);
    // MLP branch
    ln_kernel<<<4096, 256, 0, stream>>>(h, ln2g + l * 768, ln2b + l * 768, m_in);
    gemm_bf16_kernel<<<gH, 256, 0, stream>>>(m_in, W1T + (long)l * SW1, b1 + l * 3072,
                                             ffb, nullptr, 4096, 3072, 768, 1);
    gemm_bf16_kernel<<<gC, 256, 0, stream>>>(ffb, W2T + (long)l * SW1, b2 + l * 768,
                                             nullptr, h, 4096, 768, 3072, 3);
  }
}
